// Swin3DMHSA_67808943670044
// MI455X (gfx1250) — matrix operand order repaired and checked by reading, not compiled, not run on hardware
//
#include <hip/hip_runtime.h>
#include <hip/hip_bf16.h>

typedef __attribute__((ext_vector_type(16))) _Float16 v16h;
typedef __attribute__((ext_vector_type(8)))  _Float16 v8h;
typedef __attribute__((ext_vector_type(4)))  _Float16 v4h;
typedef __attribute__((ext_vector_type(8)))  float    v8f;

#define DIMX    128
#define NHEADS  4
#define NTOK    392
#define NPAD    416      // 26*16 = 13*32, padded token count
#define DH      32
#define NBATCH  256
#define TSIZE   2535     // (2*8-1)*(2*7-1)*(2*7-1)
#define NROWBLK ((NBATCH * NTOK) / 64)   // 1568, exact

// ---------------------------------------------------------------------------
// WMMA fragment loaders (wave32), A/B in contiguous k order, not the ISA table order:
//   A 16x32 f16 : lane m=lane&15, half=lane>>4; vgpr 0-3 -> K = half*16 + 2j(+1),
//                 vgpr 4-7 -> K = half*16 + 8 + 2j(+1)
//   B 32x16 f16 : lane n=lane&15, half=lane>>4; vgpr j -> K = half*16 + 2j(+1)
//   C/D 16x16 f32: lane n=lane&15, half=lane>>4; vgpr j -> row M = half*8 + j
// ---------------------------------------------------------------------------

__device__ __forceinline__ v16h load_frag_a(const _Float16* p, int row0, int k0, int ld) {
  int lane = threadIdx.x & 31;
  int m = lane & 15, hf = lane >> 4;
  const _Float16* r = p + (row0 + m) * ld + k0;
  v16h a;
#pragma unroll
  for (int j = 0; j < 4; ++j) {
    int k = hf * 16 + 2 * j;
    a[2 * j]     = r[k];
    a[2 * j + 1] = r[k + 1];
  }
#pragma unroll
  for (int j = 0; j < 4; ++j) {
    int k = hf * 16 + 8 + 2 * j;
    a[8 + 2 * j] = r[k];
    a[9 + 2 * j] = r[k + 1];
  }
  return a;
}

// B[k][n] = src[(n0+n)*ld + k0+k]  (i.e. B = src^T, src row-major N x K).
__device__ __forceinline__ v16h load_frag_bt(const _Float16* p, int n0, int k0, int ld) {
  int lane = threadIdx.x & 31;
  int n = lane & 15, hf = lane >> 4;
  const _Float16* r = p + (n0 + n) * ld + k0 + hf * 16;
  v16h b;
#pragma unroll
  for (int j = 0; j < 16; ++j) b[j] = r[j];
  return b;
}

// B[k][n] = src[(k0+k)*ld + n0+n]  (src row-major K x N).
__device__ __forceinline__ v16h load_frag_b(const _Float16* p, int k0, int n0, int ld) {
  int lane = threadIdx.x & 31;
  int n = lane & 15, hf = lane >> 4;
  v16h b;
#pragma unroll
  for (int j = 0; j < 8; ++j) {
    int k = k0 + hf * 16 + 2 * j;
    b[2 * j]     = p[k * ld + n0 + n];
    b[2 * j + 1] = p[(k + 1) * ld + n0 + n];
  }
  return b;
}

// Async global->LDS copy helpers (CDNA5, tracked by ASYNCcnt).
// ldsoff = byte offset from the wave's LDS base (dynamic LDS starts at 0 when
// no static __shared__ is used); gaddr = 64-bit global address (GV mode).
__device__ __forceinline__ void async_load_b128(unsigned ldsoff, const void* gaddr) {
  asm volatile("global_load_async_to_lds_b128 %0, %1, off"
               :: "v"(ldsoff), "v"((unsigned long long)(size_t)gaddr) : "memory");
}
__device__ __forceinline__ void async_load_b32(unsigned ldsoff, const void* gaddr) {
  asm volatile("global_load_async_to_lds_b32 %0, %1, off"
               :: "v"(ldsoff), "v"((unsigned long long)(size_t)gaddr) : "memory");
}
__device__ __forceinline__ void wait_async0() {
  asm volatile("s_wait_asynccnt 0" ::: "memory");
}

// ---------------------------------------------------------------------------
// Kernel 1: qkv = X @ Wqkv^T + b, written f16 into (b, which, h, n, dh) layout.
// Block: 256 threads (8 waves), 64 rows x 384 cols per block, Wqkv in LDS.
// ---------------------------------------------------------------------------
__global__ void __launch_bounds__(256) qkv_gemm_kernel(
    const float* __restrict__ X, const float* __restrict__ W,
    const float* __restrict__ bias, _Float16* __restrict__ qkv) {
  extern __shared__ __align__(16) char smem[];
  _Float16* xs = (_Float16*)smem;               // 64 x 128
  _Float16* ws = xs + 64 * DIMX;                // 384 x 128
  float*    bs = (float*)(ws + 3 * DIMX * DIMX);// 384

  const int tid = threadIdx.x;
  const int rowbase = blockIdx.x * 64;

  for (int i = tid; i < 3 * DIMX * DIMX; i += 256) ws[i] = (_Float16)W[i];
  for (int i = tid; i < 64 * DIMX; i += 256)       xs[i] = (_Float16)X[rowbase * DIMX + i];
  for (int i = tid; i < 3 * DIMX; i += 256)        bs[i] = bias[i];
  __syncthreads();

  const int wave = tid >> 5, lane = tid & 31;
  const int n16 = lane & 15, hf = lane >> 4;

  for (int t = wave; t < 96; t += 8) {          // 4 row-tiles x 24 col-tiles
    int rt = t / 24, ct = t % 24;
    v8f acc = {};
#pragma unroll
    for (int k = 0; k < 4; ++k) {
      v16h a = load_frag_a(xs, rt * 16, k * 32, DIMX);
      v16h b = load_frag_bt(ws, ct * 16, k * 32, DIMX);
      acc = __builtin_amdgcn_wmma_f32_16x16x32_f16(false, a, false, b,
                                                   (short)0, acc, false, false);
    }
    int col = ct * 16 + n16;
    float bv = bs[col];
    int which = col >> 7, rem = col & 127, h = rem >> 5, d = rem & 31;
    // One window-decomposition per tile, increment-with-wrap for the 8 rows.
    int r0 = rowbase + rt * 16 + hf * 8;
    int bwin = r0 / NTOK;
    int n = r0 - bwin * NTOK;
    int idx = ((bwin * 12 + which * NHEADS + h) * NTOK + n) * DH + d;
#pragma unroll
    for (int j = 0; j < 8; ++j) {
      qkv[idx] = (_Float16)(acc[j] + bv);
      ++n; idx += DH;
      if (n == NTOK) {
        n = 0; ++bwin;
        idx = ((bwin * 12 + which * NHEADS + h) * NTOK) * DH + d;
      }
    }
  }
}

// ---------------------------------------------------------------------------
// Kernel 2: fused windowed attention per (window, head).
// Block: 128 threads (4 waves).  Q/K/V (padded 416x32 f16) + rpb row staged in
// LDS via async global->LDS copies.  Each wave owns 16 query rows at a time:
// scores via WMMA, bias gather, shfl-reduced softmax (denominator folded into
// the ctx accumulator), then P@V WMMA.
// ---------------------------------------------------------------------------
__global__ void __launch_bounds__(128) attn_kernel(
    const _Float16* __restrict__ qkv, const float* __restrict__ rpb,
    const int* __restrict__ rel, _Float16* __restrict__ ctx) {
  extern __shared__ __align__(16) char smem[];
  _Float16* qs = (_Float16*)smem;               // 416 x 32
  _Float16* ks = qs + NPAD * DH;
  _Float16* vs = ks + NPAD * DH;
  _Float16* pb = vs + NPAD * DH;                // 4 waves x 16 x 416
  float*    tb = (float*)(pb + 4 * 16 * NPAD);  // 2535

  const int bh = blockIdx.x;
  const int bwin = bh >> 2, h = bh & 3;
  const int tid = threadIdx.x, wave = tid >> 5, lane = tid & 31;
  const int n16 = lane & 15, hf = lane >> 4;

  const _Float16* qg = qkv + ((bwin * 3 + 0) * NHEADS + h) * NTOK * DH;
  const _Float16* kg = qg + NHEADS * NTOK * DH;
  const _Float16* vg = qg + 2 * NHEADS * NTOK * DH;

  // Async-stage Q/K/V (16B chunks) and the head's bias-table row.
  for (int i = tid; i < (NTOK * DH) / 8; i += 128) {
    int gi = i * 8;                              // element offset, 16B aligned
    async_load_b128((unsigned)((char*)(qs + gi) - smem), qg + gi);
    async_load_b128((unsigned)((char*)(ks + gi) - smem), kg + gi);
    async_load_b128((unsigned)((char*)(vs + gi) - smem), vg + gi);
  }
  for (int i = tid; i < TSIZE; i += 128)
    async_load_b32((unsigned)((char*)(tb + i) - smem), rpb + h * TSIZE + i);
  // Zero-fill padded rows 392..415 with plain DS stores.
  for (int i = tid; i < (NPAD - NTOK) * DH / 4; i += 128) {
    v4h z = {};
    *(v4h*)(qs + NTOK * DH + i * 4) = z;
    *(v4h*)(ks + NTOK * DH + i * 4) = z;
    *(v4h*)(vs + NTOK * DH + i * 4) = z;
  }
  wait_async0();
  __syncthreads();

  _Float16* pbw = pb + wave * 16 * NPAD;
  const float SCALE = 0.17677669529663687f;     // 1/sqrt(32)

  for (int rt = wave; rt < 25; rt += 4) {       // 25 row tiles cover 392 rows
    v16h aq = load_frag_a(qs, rt * 16, 0, DH);

    float m8[8];
#pragma unroll
    for (int j = 0; j < 8; ++j) m8[j] = -3.0e30f;

    // Pass 1: scores + bias -> LDS (f16), track row maxima.
    for (int ct = 0; ct < 26; ++ct) {
      v16h bk = load_frag_bt(ks, ct * 16, 0, DH);
      v8f s = {};
      s = __builtin_amdgcn_wmma_f32_16x16x32_f16(false, aq, false, bk,
                                                 (short)0, s, false, false);
      int col = ct * 16 + n16;
#pragma unroll
      for (int j = 0; j < 8; ++j) {
        int row = rt * 16 + hf * 8 + j;
        float bias;
        if (col < NTOK)
          bias = (row < NTOK) ? tb[rel[row * NTOK + col]] : 0.0f;
        else
          bias = -30000.0f;                     // kill padded key columns
        float val = s[j] * SCALE + bias;
        m8[j] = fmaxf(m8[j], val);
        pbw[(hf * 8 + j) * NPAD + col] = (_Float16)val;
      }
    }
    // Row-max across the 16 lanes sharing each row (xor stays in half-group).
#pragma unroll
    for (int off = 8; off; off >>= 1)
#pragma unroll
      for (int j = 0; j < 8; ++j)
        m8[j] = fmaxf(m8[j], __shfl_xor(m8[j], off, 32));
    __builtin_amdgcn_wave_barrier();

    // Pass 2: exponentiate in place, accumulate row sums.
    float s8[8];
#pragma unroll
    for (int j = 0; j < 8; ++j) s8[j] = 0.0f;
    for (int ct = 0; ct < 26; ++ct) {
      int col = ct * 16 + n16;
#pragma unroll
      for (int j = 0; j < 8; ++j) {
        int o = (hf * 8 + j) * NPAD + col;
        float e = __expf((float)pbw[o] - m8[j]);
        s8[j] += e;
        pbw[o] = (_Float16)e;
      }
    }
#pragma unroll
    for (int off = 8; off; off >>= 1)
#pragma unroll
      for (int j = 0; j < 8; ++j)
        s8[j] += __shfl_xor(s8[j], off, 32);
    __builtin_amdgcn_wave_barrier();
    asm volatile("s_wait_dscnt 0" ::: "memory");

    // ctx = P @ V ; softmax denominator folded into the accumulator
    // (D-row (hf*8+j) has the same lane mapping as P-row).
    v8f acc = {};
#pragma unroll
    for (int kt = 0; kt < 13; ++kt) {
      v16h ap = load_frag_a(pbw, 0, kt * 32, NPAD);
      v16h bv = load_frag_b(vs, kt * 32, 0, DH);
      acc = __builtin_amdgcn_wmma_f32_16x16x32_f16(false, ap, false, bv,
                                                   (short)0, acc, false, false);
    }
#pragma unroll
    for (int j = 0; j < 8; ++j) {
      int row = rt * 16 + hf * 8 + j;
      if (row < NTOK)
        ctx[(bwin * NTOK + row) * DIMX + h * DH + n16] =
            (_Float16)(acc[j] / s8[j]);
    }
  }
}

// ---------------------------------------------------------------------------
// Kernel 3: out = ctx @ proj_w^T + proj_b  (f16 in, f32 out).
// ---------------------------------------------------------------------------
__global__ void __launch_bounds__(256) proj_gemm_kernel(
    const _Float16* __restrict__ ctx, const float* __restrict__ W,
    const float* __restrict__ bias, float* __restrict__ out) {
  extern __shared__ __align__(16) char smem[];
  _Float16* cs  = (_Float16*)smem;              // 64 x 128
  _Float16* wsm = cs + 64 * DIMX;               // 128 x 128
  float*    bs  = (float*)(wsm + DIMX * DIMX);  // 128

  const int tid = threadIdx.x;
  const int rowbase = blockIdx.x * 64;

  for (int i = tid; i < DIMX * DIMX; i += 256) wsm[i] = (_Float16)W[i];
  for (int i = tid; i < 64 * DIMX / 8; i += 256)
    async_load_b128((unsigned)((char*)(cs + i * 8) - smem),
                    ctx + rowbase * DIMX + i * 8);
  for (int i = tid; i < DIMX; i += 256) bs[i] = bias[i];
  wait_async0();
  __syncthreads();

  const int wave = tid >> 5, lane = tid & 31;
  const int n16 = lane & 15, hf = lane >> 4;

  for (int t = wave; t < 32; t += 8) {          // 4 row-tiles x 8 col-tiles
    int rt = t >> 3, ct = t & 7;
    v8f acc = {};
#pragma unroll
    for (int k = 0; k < 4; ++k) {
      v16h a = load_frag_a(cs, rt * 16, k * 32, DIMX);
      v16h b = load_frag_bt(wsm, ct * 16, k * 32, DIMX);
      acc = __builtin_amdgcn_wmma_f32_16x16x32_f16(false, a, false, b,
                                                   (short)0, acc, false, false);
    }
    int col = ct * 16 + n16;
    float bv = bs[col];
#pragma unroll
    for (int j = 0; j < 8; ++j) {
      int r = rowbase + rt * 16 + hf * 8 + j;
      out[r * DIMX + col] = acc[j] + bv;
    }
  }
}

// ---------------------------------------------------------------------------
extern "C" void kernel_launch(void* const* d_in, const int* in_sizes, int n_in,
                              void* d_out, int out_size, void* d_ws, size_t ws_size,
                              hipStream_t stream) {
  const float* X    = (const float*)d_in[0];   // (256, 8,7,7, 128)
  const float* Wqkv = (const float*)d_in[1];   // (384, 128)
  const float* bqkv = (const float*)d_in[2];   // (384,)
  const float* Wp   = (const float*)d_in[3];   // (128, 128)
  const float* bp   = (const float*)d_in[4];   // (128,)
  const float* rpb  = (const float*)d_in[5];   // (4, 2535)
  const int*   rel  = (const int*)d_in[6];     // (392*392,)
  float* out = (float*)d_out;

  _Float16* qkv = (_Float16*)d_ws;                                  // 77.1 MB
  _Float16* ctx = qkv + (size_t)NBATCH * 3 * NHEADS * NTOK * DH;    // 25.7 MB

  const int lds1 = (64 * DIMX + 3 * DIMX * DIMX) * 2 + 3 * DIMX * 4;        // 116224
  const int lds2 = (3 * NPAD * DH + 4 * 16 * NPAD) * 2 + TSIZE * 4;        // 143260
  const int lds3 = (64 * DIMX + DIMX * DIMX) * 2 + DIMX * 4;               // 49664

  qkv_gemm_kernel<<<NROWBLK, 256, lds1, stream>>>(X, Wqkv, bqkv, qkv);
  attn_kernel<<<NBATCH * NHEADS, 128, lds2, stream>>>(qkv, rpb, rel, ctx);
  proj_gemm_kernel<<<NROWBLK, 256, lds3, stream>>>(ctx, Wp, bp, out);
}
